// SpatialAttention_w_skip_25692494365578
// MI455X (gfx1250) — compile-verified, matrix operand order checked by reading, not run on hardware
//
#include <hip/hip_runtime.h>
#include <hip/hip_bf16.h>

// ---------------------------------------------------------------------------
// SpatialAttention with global softmax, fused flash-style on CDNA5 (gfx1250)
// B=8, C_IN=256, C_OUT=256, QK_C=32, N=H*W=4096
// ---------------------------------------------------------------------------

#define BATCH 8
#define CIN   256
#define COUT  256
#define QKC   32
#define NSP   4096   // 64*64

#define LDS_ROW 40   // bf16 elems per LDS row (32 data + 8 pad = 80B, bank-friendly)

typedef __bf16 bf16_t;
typedef bf16_t v16bf __attribute__((ext_vector_type(16)));
typedef bf16_t v8bf  __attribute__((ext_vector_type(8)));
typedef float  v8f   __attribute__((ext_vector_type(8)));
typedef float  v4f   __attribute__((ext_vector_type(4)));

static __device__ __forceinline__ v8f wmma_bf16(v16bf a, v16bf b, v8f c) {
  // D = A(16x32 bf16) x B(32x16 bf16) + C(16x16 f32)
  return __builtin_amdgcn_wmma_f32_16x16x32_bf16(
      /*neg_a=*/false, a, /*neg_b=*/false, b,
      /*c_mod=*/(short)0, c, /*reuse_a=*/false, /*reuse_b=*/false);
}

// Load one 16x32 bf16 WMMA operand (A-layout; B lane-symmetric) from a
// row-major [row][row_stride] bf16 source. Lane l: row = l&15,
// elements 0..7  <- cols (l<16 ? 0 : 8)  .. +7
// elements 8..15 <- cols (l<16 ? 16: 24) .. +7   (two 16B loads per lane)
static __device__ __forceinline__ v16bf load_op(const bf16_t* __restrict__ base,
                                                int row_stride, int lane) {
  const bf16_t* p = base + (size_t)(lane & 15) * row_stride + ((lane >> 4) << 3);
  v8bf lo = *reinterpret_cast<const v8bf*>(p);
  v8bf hi = *reinterpret_cast<const v8bf*>(p + 16);
  return __builtin_shufflevector(lo, hi, 0,1,2,3,4,5,6,7,8,9,10,11,12,13,14,15);
}

// Same operand fetch, but from an LDS tile with LDS_ROW-element padded rows.
static __device__ __forceinline__ v16bf load_op_lds(const bf16_t* base, int lane) {
  const bf16_t* p = base + (lane & 15) * LDS_ROW + ((lane >> 4) << 3);
  v8bf lo = *reinterpret_cast<const v8bf*>(p);
  v8bf hi = *reinterpret_cast<const v8bf*>(p + 16);
  return __builtin_shufflevector(lo, hi, 0,1,2,3,4,5,6,7,8,9,10,11,12,13,14,15);
}

// Async DMA one 64B row chunk (global -> LDS), 4x b128; tracked by ASYNCcnt.
// INST_OFFSET applies to both the LDS and the global address (ISA 10.x async).
static __device__ __forceinline__ void stage_row_async(const bf16_t* gsrc, unsigned lds_off) {
  unsigned long long ga = (unsigned long long)(uintptr_t)gsrc;
  asm volatile(
      "global_load_async_to_lds_b128 %0, %1, off\n\t"
      "global_load_async_to_lds_b128 %0, %1, off offset:16\n\t"
      "global_load_async_to_lds_b128 %0, %1, off offset:32\n\t"
      "global_load_async_to_lds_b128 %0, %1, off offset:48"
      :: "v"(lds_off), "v"(ga) : "memory");
}

// ---------------------------------------------------------------------------
// K0a: x [B][C][N] f32  ->  xbT [B][N][C] bf16   (LDS tiled transpose)
// ---------------------------------------------------------------------------
__global__ void cvt_x_kernel(const float* __restrict__ x, bf16_t* __restrict__ xbT) {
  __shared__ float tile[16][17];
  int b  = blockIdx.x;
  int c0 = blockIdx.y * 16;
  int n0 = blockIdx.z * 16;
  int tx = threadIdx.x, ty = threadIdx.y;
  tile[ty][tx] = x[((size_t)b * CIN + c0 + ty) * NSP + n0 + tx];
  __syncthreads();
  xbT[((size_t)b * NSP + n0 + ty) * CIN + c0 + tx] = (bf16_t)tile[tx][ty];
}

// K0b: f32 -> bf16 flat convert (weights)
__global__ void cvt_w_kernel(const float* __restrict__ src, bf16_t* __restrict__ dst, int n) {
  int i = blockIdx.x * blockDim.x + threadIdx.x;
  if (i < n) dst[i] = (bf16_t)src[i];
}

// ---------------------------------------------------------------------------
// K1: qkv GEMM.  D[n][o] = sum_c xbT[n][c] * W[o][c]  + bias[o]
//   qk_mode==0: store vb[b][o][n] bf16 (packed 16B stores)
//   qk_mode==1: store qT/kT[b][n][d] bf16
// ---------------------------------------------------------------------------
__global__ void __launch_bounds__(128)
qkv_gemm_kernel(const bf16_t* __restrict__ xbT, const bf16_t* __restrict__ Wb,
                const float* __restrict__ bias, bf16_t* __restrict__ dst, int qk_mode) {
  const int b    = blockIdx.x;
  const int wave = threadIdx.x >> 5;
  const int lane = threadIdx.x & 31;
  const int n0   = (blockIdx.y * 4 + wave) * 64;
  const int dt   = blockIdx.z * 16;

  const v8f vz = {0.f,0.f,0.f,0.f,0.f,0.f,0.f,0.f};
  v8f acc[4] = {vz, vz, vz, vz};

  #pragma unroll
  for (int kc = 0; kc < 8; ++kc) {
    const int c0 = kc * 32;
    v16bf bw = load_op(Wb + (size_t)dt * CIN + c0, CIN, lane);
    #pragma unroll
    for (int i = 0; i < 4; ++i) {
      v16bf a = load_op(xbT + ((size_t)b * NSP + n0 + i * 16) * CIN + c0, CIN, lane);
      acc[i] = wmma_bf16(a, bw, acc[i]);
    }
  }

  const float bval = bias[dt + (lane & 15)];
  if (qk_mode == 0) {
    bf16_t* vrow = dst + ((size_t)b * COUT + dt + (lane & 15)) * NSP;
    #pragma unroll
    for (int i = 0; i < 4; ++i) {
      v8bf p;
      #pragma unroll
      for (int r = 0; r < 8; ++r) p[r] = (bf16_t)(acc[i][r] + bval);
      *reinterpret_cast<v8bf*>(vrow + n0 + i * 16 + ((lane >> 4) << 3)) = p;
    }
  } else {
    bf16_t* qrow = dst + (size_t)b * NSP * QKC;
    const int d = dt + (lane & 15);
    #pragma unroll
    for (int i = 0; i < 4; ++i) {
      #pragma unroll
      for (int r = 0; r < 8; ++r) {
        int n = n0 + i * 16 + r + ((lane >> 4) << 3);
        qrow[(size_t)n * QKC + d] = (bf16_t)(acc[i][r] + bval);
      }
    }
  }
}

// ---------------------------------------------------------------------------
// K2: online global-softmax statistics.  S[n][m] = sum_d qT[n][d]*kT[m][d]
// ---------------------------------------------------------------------------
__global__ void __launch_bounds__(256)
score_stats_kernel(const bf16_t* __restrict__ qT, const bf16_t* __restrict__ kT,
                   float* __restrict__ Lmax, float* __restrict__ Lsum) {
  const int b    = blockIdx.x;
  const int mt   = blockIdx.y * 8 + (threadIdx.x >> 5);
  const int lane = threadIdx.x & 31;
  const v8f vz = {0.f,0.f,0.f,0.f,0.f,0.f,0.f,0.f};

  v16bf bk = load_op(kT + ((size_t)b * NSP + mt * 16) * QKC, QKC, lane);

  float mloc = -1e30f, sloc = 0.f;
  for (int nt = 0; nt < NSP / 16; ++nt) {
    v16bf aq = load_op(qT + ((size_t)b * NSP + nt * 16) * QKC, QKC, lane);
    v8f s = wmma_bf16(aq, bk, vz);
    float tmax = s[0];
    #pragma unroll
    for (int j = 1; j < 8; ++j) tmax = fmaxf(tmax, s[j]);
    float nm = fmaxf(mloc, tmax);
    sloc *= __expf(mloc - nm);
    #pragma unroll
    for (int j = 0; j < 8; ++j) sloc += __expf(s[j] - nm);
    mloc = nm;
  }
  #pragma unroll
  for (int off = 16; off >= 1; off >>= 1) {
    float om = __shfl_xor(mloc, off, 32);
    float os = __shfl_xor(sloc, off, 32);
    float nm = fmaxf(mloc, om);
    sloc = sloc * __expf(mloc - nm) + os * __expf(om - nm);
    mloc = nm;
  }
  if (lane == 0) {
    Lmax[b * 256 + mt] = mloc;
    Lsum[b * 256 + mt] = sloc;
  }
}

// K3: combine 256 per-tile (max,sum) partials per batch -> mZ[b] = {M, Z}
__global__ void __launch_bounds__(256)
stats_reduce_kernel(const float* __restrict__ Lmax, const float* __restrict__ Lsum,
                    float* __restrict__ mZ) {
  __shared__ float sm[256];
  __shared__ float ss[256];
  const int b = blockIdx.x, t = threadIdx.x;
  sm[t] = Lmax[b * 256 + t];
  ss[t] = Lsum[b * 256 + t];
  __syncthreads();
  for (int s = 128; s > 0; s >>= 1) {
    if (t < s) {
      float nm = fmaxf(sm[t], sm[t + s]);
      ss[t] = ss[t] * __expf(sm[t] - nm) + ss[t + s] * __expf(sm[t + s] - nm);
      sm[t] = nm;
    }
    __syncthreads();
  }
  if (t == 0) { mZ[b * 2] = sm[0]; mZ[b * 2 + 1] = ss[0]; }
}

// ---------------------------------------------------------------------------
// K4: fused output.  out^T[m][o] = sum_n exp(S[n][m]-M) * v[o][n];
// final out[o][m] = out^T * (1/Z) + x[o][m].
// 8 waves/workgroup share one batch; the 16KB v-tile [256 o][32 n] per n-step
// is identical for all 8 waves, so it is staged ONCE per workgroup into LDS
// with async global->LDS DMA (double-buffered, overlapped with the 144 WMMAs
// of the previous step), cutting L2 traffic 8x. Waves read B-operands back
// with ds_load_b128 from 80B-padded rows (conflict-free bank groups).
// ---------------------------------------------------------------------------
__global__ void __launch_bounds__(256)
out_fused_kernel(const bf16_t* __restrict__ qT, const bf16_t* __restrict__ kT,
                 const bf16_t* __restrict__ vb, const float* __restrict__ mZ,
                 const float* __restrict__ x, float* __restrict__ out) {
  __shared__ __align__(16) bf16_t vbs[2][256 * LDS_ROW];  // 2 x 20KB

  const int b    = blockIdx.x;
  const int tid  = threadIdx.x;
  const int mt   = blockIdx.y * 8 + (tid >> 5);
  const int lane = tid & 31;
  const v8f vz = {0.f,0.f,0.f,0.f,0.f,0.f,0.f,0.f};

  const float M    = mZ[b * 2];
  const float invZ = 1.f / mZ[b * 2 + 1];

  v16bf bk = load_op(kT + ((size_t)b * NSP + mt * 16) * QKC, QKC, lane);

  v8f acc[16];
  #pragma unroll
  for (int i = 0; i < 16; ++i) acc[i] = vz;

  // Each thread stages one v-row (64B) per step: row o = tid, cols [n0, n0+32)
  const bf16_t* vrow_g = vb + ((size_t)b * COUT + tid) * NSP;
  const unsigned lds_row_off[2] = {
      (unsigned)(unsigned long long)(__attribute__((address_space(3))) bf16_t*)&vbs[0][tid * LDS_ROW],
      (unsigned)(unsigned long long)(__attribute__((address_space(3))) bf16_t*)&vbs[1][tid * LDS_ROW]};

  constexpr int T = NSP / 32;
  stage_row_async(vrow_g, lds_row_off[0]);          // prologue: tile 0

  for (int t = 0; t < T; ++t) {
    const int n0 = t * 32;
    if (t + 1 < T) {
      stage_row_async(vrow_g + (t + 1) * 32, lds_row_off[(t + 1) & 1]);
      asm volatile("s_wait_asynccnt 0x4" ::: "memory");   // tile t landed
    } else {
      asm volatile("s_wait_asynccnt 0x0" ::: "memory");
    }
    __syncthreads();                                 // publish tile t

    v16bf aq0 = load_op(qT + ((size_t)b * NSP + n0) * QKC, QKC, lane);
    v16bf aq1 = load_op(qT + ((size_t)b * NSP + n0 + 16) * QKC, QKC, lane);
    v8f s0 = wmma_bf16(aq0, bk, vz);
    v8f s1 = wmma_bf16(aq1, bk, vz);
    // P^T A-operand: lane-local (f32 D layout == bf16 A layout per half)
    v16bf pa;
    #pragma unroll
    for (int j = 0; j < 8; ++j) {
      pa[j]     = (bf16_t)__expf(s0[j] - M);
      pa[j + 8] = (bf16_t)__expf(s1[j] - M);
    }
    const bf16_t* tile = &vbs[t & 1][0];
    #pragma unroll
    for (int ot = 0; ot < 16; ++ot) {
      v16bf bv = load_op_lds(tile + (ot * 16) * LDS_ROW, lane);
      acc[ot] = wmma_bf16(pa, bv, acc[ot]);
    }
    __syncthreads();                                 // done reading tile t
  }

  // Epilogue: acc[ot]: N=o = ot*16+(lane&15); M=m = mt*16 + r + 8*(lane>>4)
  const int mbase = mt * 16 + ((lane >> 4) << 3);
  #pragma unroll
  for (int ot = 0; ot < 16; ++ot) {
    const int o = ot * 16 + (lane & 15);
    const size_t idx = ((size_t)b * COUT + o) * NSP + mbase;
    const v4f xa = *reinterpret_cast<const v4f*>(x + idx);
    const v4f xb = *reinterpret_cast<const v4f*>(x + idx + 4);
    v4f r0, r1;
    #pragma unroll
    for (int j = 0; j < 4; ++j) {
      r0[j] = acc[ot][j]     * invZ + xa[j];
      r1[j] = acc[ot][j + 4] * invZ + xb[j];
    }
    *reinterpret_cast<v4f*>(out + idx)     = r0;
    *reinterpret_cast<v4f*>(out + idx + 4) = r1;
  }
}

// ---------------------------------------------------------------------------
// Host launcher
// ---------------------------------------------------------------------------
extern "C" void kernel_launch(void* const* d_in, const int* in_sizes, int n_in,
                              void* d_out, int out_size, void* d_ws, size_t ws_size,
                              hipStream_t stream) {
  const float* x  = (const float*)d_in[0];
  const float* W1 = (const float*)d_in[1];
  const float* b1 = (const float*)d_in[2];
  const float* W2 = (const float*)d_in[3];
  const float* b2 = (const float*)d_in[4];
  const float* W3 = (const float*)d_in[5];
  const float* b3 = (const float*)d_in[6];
  float* out = (float*)d_out;

  char* ws = (char*)d_ws;
  size_t off = 0;
  auto alloc = [&](size_t bytes) { char* p = ws + off; off = (off + bytes + 255) & ~(size_t)255; return p; };

  bf16_t* xbT = (bf16_t*)alloc((size_t)BATCH * NSP * CIN * 2);   // 16 MB
  bf16_t* Wb1 = (bf16_t*)alloc((size_t)COUT * CIN * 2);
  bf16_t* Wb2 = (bf16_t*)alloc((size_t)QKC * CIN * 2);
  bf16_t* Wb3 = (bf16_t*)alloc((size_t)QKC * CIN * 2);
  bf16_t* vb  = (bf16_t*)alloc((size_t)BATCH * COUT * NSP * 2);  // 16 MB
  bf16_t* qT  = (bf16_t*)alloc((size_t)BATCH * NSP * QKC * 2);   // 2 MB
  bf16_t* kT  = (bf16_t*)alloc((size_t)BATCH * NSP * QKC * 2);   // 2 MB
  float*  Lmax = (float*)alloc((size_t)BATCH * 256 * 4);
  float*  Lsum = (float*)alloc((size_t)BATCH * 256 * 4);
  float*  mZ   = (float*)alloc((size_t)BATCH * 2 * 4);
  (void)ws_size; (void)in_sizes; (void)n_in; (void)out_size;

  // K0: convert / transpose inputs to bf16
  cvt_x_kernel<<<dim3(BATCH, CIN / 16, NSP / 16), dim3(16, 16), 0, stream>>>(x, xbT);
  cvt_w_kernel<<<dim3((COUT * CIN + 255) / 256), 256, 0, stream>>>(W1, Wb1, COUT * CIN);
  cvt_w_kernel<<<dim3((QKC * CIN + 255) / 256), 256, 0, stream>>>(W2, Wb2, QKC * CIN);
  cvt_w_kernel<<<dim3((QKC * CIN + 255) / 256), 256, 0, stream>>>(W3, Wb3, QKC * CIN);

  // K1: v / q / k projections (WMMA)
  qkv_gemm_kernel<<<dim3(BATCH, 16, COUT / 16), 128, 0, stream>>>(xbT, Wb1, b1, vb, 0);
  qkv_gemm_kernel<<<dim3(BATCH, 16, QKC / 16), 128, 0, stream>>>(xbT, Wb2, b2, qT, 1);
  qkv_gemm_kernel<<<dim3(BATCH, 16, QKC / 16), 128, 0, stream>>>(xbT, Wb3, b3, kT, 1);

  // K2+K3: global-softmax statistics (scores never materialized)
  score_stats_kernel<<<dim3(BATCH, 32), 256, 0, stream>>>(qT, kT, Lmax, Lsum);
  stats_reduce_kernel<<<dim3(BATCH), 256, 0, stream>>>(Lmax, Lsum, mZ);

  // K4: fused  out = v * exp(S - M)/Z + x   (async-LDS staged v-tiles)
  out_fused_kernel<<<dim3(BATCH, 32), 256, 0, stream>>>(qT, kT, vb, mZ, x, out);
}
